// PrimaryKeyGNN_13048110645453
// MI455X (gfx1250) — compile-verified, matrix operand order checked by reading, not run on hardware
//
#include <hip/hip_runtime.h>
#include <hip/hip_bf16.h>
#include <math.h>

// ---------------------------------------------------------------------------
// GAT (2-layer) for MI455X / gfx1250, wave32.
//   N=50000 nodes, E=800000 edges (+N self loops), IN=128, H1=4, HID=64, OUT=64
// GEMMs: v_wmma_f32_16x16x32_f16, B panel staged in LDS in fragment order
// (ds_load_b128 per fragment), 2 output tiles per wave sharing one A fragment,
// A loaded as float4 (global_load_b128) and packed to f16 in-register.
// Edge softmax/aggregation: 3-pass segment softmax with float atomics,
// channel-parallel coalesced gather/scatter (memory-bound part, ~1.7GB).
// ---------------------------------------------------------------------------

typedef __attribute__((ext_vector_type(16))) _Float16 v16h;
typedef __attribute__((ext_vector_type(8)))  float    v8f;

#define K_IN   128
#define HC1    256   // H1*HID
#define H1_    4
#define HID_   64
#define OUT_   64

// ---------------- weight f32 -> f16 conversion ----------------
__global__ void cvt_f16_kernel(const float* __restrict__ w, _Float16* __restrict__ o, int n) {
    int i = blockIdx.x * blockDim.x + threadIdx.x;
    if (i < n) o[i] = (_Float16)w[i];
}

// ---------------- fill ----------------
__global__ void fill_kernel(float* __restrict__ p, float v, int n) {
    int i = blockIdx.x * blockDim.x + threadIdx.x;
    for (; i < n; i += gridDim.x * blockDim.x) p[i] = v;
}

// ---------------- GEMM: C[M,NO] = A_f32[M,K] @ B_f16[K,NO], f32 accum ------
// Fragment layouts (ISA 7.12.2, 16-bit 16x32 A / 32x16 B, wave32):
//   lane = loc%16 + 16*half ; VGPR v holds K pair at
//   koff(v,half) = (v<4?0:16) + (v&3)*2 + half*8
// B is staged in LDS pre-swizzled into exact fragment order so each lane
// fetches its 16 f16 (32B) contiguously (2x ds_load_b128).
// Each wave computes TWO adjacent 16x16 column tiles sharing one A fragment.
template <int K, int NO>
__global__ void gemm_wmma_f16(const float* __restrict__ A,
                              const _Float16* __restrict__ B,
                              float* __restrict__ C,
                              int totalPairs) {
    constexpr int NT = NO >> 4;          // 16-wide column tiles
    constexpr int KB = K >> 5;           // 32-deep K blocks
    __shared__ _Float16 ldsB[K * NO];    // 64KB (layer1) / 32KB (layer2)

    const int tid = threadIdx.x;
    // ---- cooperative fill: coalesced global read, fragment-order LDS write
    for (int idx = tid; idx < K * NO; idx += blockDim.x) {
        const int k    = idx / NO;       // compile-time NO -> shifts
        const int col  = idx - k * NO;
        const int kblk = k >> 5;
        const int ko   = k & 31;
        const int hf   = (ko >> 3) & 1;
        const int v    = ((ko & 16) >> 2) + ((ko & 7) >> 1);  // (ko>=16?4:0)+((ko&7)>>1)
        const int pr   = ko & 1;
        const int tn   = col >> 4;
        const int nloc = col & 15;
        const int ln   = hf * 16 + nloc;
        ldsB[((kblk * NT + tn) * 32 + ln) * 16 + 2 * v + pr] = B[idx];
    }
    __syncthreads();

    const int wave = tid >> 5;
    const int lane = tid & 31;
    const int half = lane >> 4;
    const int mloc = lane & 15;
    constexpr int PPS = NO >> 5;         // column-tile pairs per 16-row stripe

    for (int p = blockIdx.x * (blockDim.x >> 5) + wave; p < totalPairs;
         p += gridDim.x * (blockDim.x >> 5)) {              // wave-uniform loop
        const int stripe = p / PPS;
        const int tn0 = (p - stripe * PPS) * 2;
        const float* __restrict__ arow = A + (size_t)(stripe * 16 + mloc) * K;

        v8f acc0 = {}, acc1 = {};
#pragma unroll
        for (int kblk = 0; kblk < KB; ++kblk) {
            // A fragment: two runs of 8 consecutive floats -> 4x b128 loads
            const float4 p0 = *(const float4*)(arow + kblk * 32 + half * 8);
            const float4 p1 = *(const float4*)(arow + kblk * 32 + half * 8 + 4);
            const float4 q0 = *(const float4*)(arow + kblk * 32 + 16 + half * 8);
            const float4 q1 = *(const float4*)(arow + kblk * 32 + 16 + half * 8 + 4);
            v16h a;
            a[0]  = (_Float16)p0.x; a[1]  = (_Float16)p0.y;
            a[2]  = (_Float16)p0.z; a[3]  = (_Float16)p0.w;
            a[4]  = (_Float16)p1.x; a[5]  = (_Float16)p1.y;
            a[6]  = (_Float16)p1.z; a[7]  = (_Float16)p1.w;
            a[8]  = (_Float16)q0.x; a[9]  = (_Float16)q0.y;
            a[10] = (_Float16)q0.z; a[11] = (_Float16)q0.w;
            a[12] = (_Float16)q1.x; a[13] = (_Float16)q1.y;
            a[14] = (_Float16)q1.z; a[15] = (_Float16)q1.w;

            // B fragments: 32B contiguous per lane (2x ds_load_b128 each)
            const v16h b0 = *(const v16h*)&ldsB[((kblk * NT + tn0) * 32 + lane) * 16];
            const v16h b1 = *(const v16h*)&ldsB[((kblk * NT + tn0 + 1) * 32 + lane) * 16];

            acc0 = __builtin_amdgcn_wmma_f32_16x16x32_f16(
                false, a, false, b0, (short)0, acc0, false, false);
            acc1 = __builtin_amdgcn_wmma_f32_16x16x32_f16(
                false, a, false, b1, (short)0, acc1, false, false);
        }
#pragma unroll
        for (int r = 0; r < 8; ++r) {
            const int rr = stripe * 16 + r + half * 8;
            C[(size_t)rr * NO + tn0 * 16 + mloc]       = acc0[r];
            C[(size_t)rr * NO + (tn0 + 1) * 16 + mloc] = acc1[r];
        }
    }
}

// ---------------- attention scores: al[n,h] = sum_c h[n,h,c]*a[h,c] --------
// blockDim == H*C (256 or 64), one block per node. LDS tree reduce per head.
__global__ void attn_scores(const float* __restrict__ h,
                            const float* __restrict__ a_src,
                            const float* __restrict__ a_dst,
                            float* __restrict__ al_src,
                            float* __restrict__ al_dst,
                            int H, int C) {
    __shared__ float ss[HC1];
    __shared__ float sd[HC1];
    const int node = blockIdx.x;
    const int tid  = threadIdx.x;
    const float v  = h[(size_t)node * H * C + tid];
    ss[tid] = v * a_src[tid];
    sd[tid] = v * a_dst[tid];
    __syncthreads();
    const int lane_in_head = tid % C;
    for (int s = C >> 1; s > 0; s >>= 1) {
        if (lane_in_head < s) {
            ss[tid] += ss[tid + s];
            sd[tid] += sd[tid + s];
        }
        __syncthreads();
    }
    if (lane_in_head == 0) {
        const int hd = tid / C;
        al_src[(size_t)node * H + hd] = ss[tid];
        al_dst[(size_t)node * H + hd] = sd[tid];
    }
}

// ---------------- edge pass 1: e = leaky_relu(als[src]+ald[dst]); seg max --
__global__ void edge_logits_max(const long long* __restrict__ ei, int E, int EA,
                                const float* __restrict__ al_src,
                                const float* __restrict__ al_dst,
                                float* __restrict__ ebuf,
                                float* __restrict__ m, int H) {
    const int idx = blockIdx.x * blockDim.x + threadIdx.x;
    if (idx >= EA * H) return;
    const int e  = idx / H;
    const int hd = idx - e * H;
    const int src = (e < E) ? (int)ei[e]     : (e - E);
    const int dst = (e < E) ? (int)ei[E + e] : (e - E);
    float x = al_src[(size_t)src * H + hd] + al_dst[(size_t)dst * H + hd];
    x = (x > 0.0f) ? x : 0.2f * x;                 // leaky_relu(0.2)
    ebuf[idx] = x;
    atomicMax(&m[(size_t)dst * H + hd], x);        // global_atomic_max_num_f32
}

// ---------------- edge pass 2: ex = exp(e - m[dst]); seg sum --------------
__global__ void edge_expsum(const long long* __restrict__ ei, int E, int EA,
                            float* __restrict__ ebuf,
                            const float* __restrict__ m,
                            float* __restrict__ den, int H) {
    const int idx = blockIdx.x * blockDim.x + threadIdx.x;
    if (idx >= EA * H) return;
    const int e  = idx / H;
    const int hd = idx - e * H;
    const int dst = (e < E) ? (int)ei[E + e] : (e - E);
    const float ex = __expf(ebuf[idx] - m[(size_t)dst * H + hd]);
    ebuf[idx] = ex;
    atomicAdd(&den[(size_t)dst * H + hd], ex);
}

// ---------------- edge pass 3: out[dst] += h[src] * alpha ------------------
// One block per edge, blockDim == H*C: fully coalesced gather + scatter-add.
__global__ void edge_aggregate(const long long* __restrict__ ei, int E,
                               const float* __restrict__ h,
                               const float* __restrict__ ebuf,
                               const float* __restrict__ den,
                               float* __restrict__ out, int H, int C) {
    const int e   = blockIdx.x;
    const int tid = threadIdx.x;
    const int hd  = tid / C;
    const int src = (e < E) ? (int)ei[e]     : (e - E);
    const int dst = (e < E) ? (int)ei[E + e] : (e - E);
    const float alpha = ebuf[(size_t)e * H + hd] /
                        (den[(size_t)dst * H + hd] + 1e-16f);
    const float msg = h[(size_t)src * H * C + tid] * alpha;
    atomicAdd(&out[(size_t)dst * H * C + tid], msg);
}

// ---------------- act = elu(agg + b) (in place) ----------------------------
__global__ void elu_bias(float* __restrict__ p, const float* __restrict__ b,
                         int hc, int n) {
    int i = blockIdx.x * blockDim.x + threadIdx.x;
    for (; i < n; i += gridDim.x * blockDim.x) {
        const float v = p[i] + b[i % hc];
        p[i] = (v > 0.0f) ? v : expm1f(v);
    }
}

// ---------------- out += b ------------------------------------------------
__global__ void add_bias(float* __restrict__ p, const float* __restrict__ b,
                         int hc, int n) {
    int i = blockIdx.x * blockDim.x + threadIdx.x;
    for (; i < n; i += gridDim.x * blockDim.x) p[i] += b[i % hc];
}

// ---------------------------------------------------------------------------
extern "C" void kernel_launch(void* const* d_in, const int* in_sizes, int n_in,
                              void* d_out, int out_size, void* d_ws, size_t ws_size,
                              hipStream_t stream) {
    const float*     x      = (const float*)d_in[0];
    const long long* ei     = (const long long*)d_in[1];   // int64 [2,E]
    const float*     W1     = (const float*)d_in[2];
    const float*     a_src1 = (const float*)d_in[3];
    const float*     a_dst1 = (const float*)d_in[4];
    const float*     b1     = (const float*)d_in[5];
    const float*     W2     = (const float*)d_in[6];
    const float*     a_src2 = (const float*)d_in[7];
    const float*     a_dst2 = (const float*)d_in[8];
    const float*     b2     = (const float*)d_in[9];
    float* out = (float*)d_out;

    const int N  = in_sizes[0] / K_IN;      // 50000
    const int E  = in_sizes[1] / 2;         // 800000
    const int EA = E + N;                   // with self loops

    // ---- workspace layout (256B aligned), ~136 MB total ----
    size_t off = 0;
    auto carve = [&](size_t bytes) -> char* {
        char* p = (char*)d_ws + off;
        off = (off + bytes + 255) & ~(size_t)255;
        return p;
    };
    _Float16* W1h  = (_Float16*)carve((size_t)K_IN * HC1 * 2);
    _Float16* W2h  = (_Float16*)carve((size_t)HC1 * OUT_ * 2);
    float* h1   = (float*)carve((size_t)N * HC1 * 4);
    float* agg1 = (float*)carve((size_t)N * HC1 * 4);
    float* h2   = (float*)carve((size_t)N * OUT_ * 4);
    float* als1 = (float*)carve((size_t)N * H1_ * 4);
    float* ald1 = (float*)carve((size_t)N * H1_ * 4);
    float* m1   = (float*)carve((size_t)N * H1_ * 4);
    float* dn1  = (float*)carve((size_t)N * H1_ * 4);
    float* als2 = (float*)carve((size_t)N * 4);
    float* ald2 = (float*)carve((size_t)N * 4);
    float* m2   = (float*)carve((size_t)N * 4);
    float* dn2  = (float*)carve((size_t)N * 4);
    float* eb1  = (float*)carve((size_t)EA * H1_ * 4);
    float* eb2  = (float*)carve((size_t)EA * 4);
    (void)ws_size;

    // ---- weights -> f16 ----
    cvt_f16_kernel<<<(K_IN * HC1 + 255) / 256, 256, 0, stream>>>(W1, W1h, K_IN * HC1);
    cvt_f16_kernel<<<(HC1 * OUT_ + 255) / 256, 256, 0, stream>>>(W2, W2h, HC1 * OUT_);

    // ============================ layer 1 ============================
    // h1 = x @ W1  (WMMA, B in LDS)
    {
        const int totalPairs = (N / 16) * (HC1 / 32);      // 25000
        int blocks = (totalPairs + 7) / 8;
        if (blocks > 1536) blocks = 1536;
        gemm_wmma_f16<K_IN, HC1><<<blocks, 256, 0, stream>>>(x, W1h, h1, totalPairs);
    }
    attn_scores<<<N, HC1, 0, stream>>>(h1, a_src1, a_dst1, als1, ald1, H1_, HID_);

    fill_kernel<<<2048, 256, 0, stream>>>(m1, -INFINITY, N * H1_);
    fill_kernel<<<2048, 256, 0, stream>>>(dn1, 0.0f, N * H1_);
    fill_kernel<<<4096, 256, 0, stream>>>(agg1, 0.0f, N * HC1);

    edge_logits_max<<<(EA * H1_ + 255) / 256, 256, 0, stream>>>(ei, E, EA, als1, ald1, eb1, m1, H1_);
    edge_expsum    <<<(EA * H1_ + 255) / 256, 256, 0, stream>>>(ei, E, EA, eb1, m1, dn1, H1_);
    edge_aggregate <<<EA, HC1, 0, stream>>>(ei, E, h1, eb1, dn1, agg1, H1_, HID_);

    elu_bias<<<4096, 256, 0, stream>>>(agg1, b1, HC1, N * HC1);

    // ============================ layer 2 ============================
    // h2 = act @ W2  (WMMA, B in LDS)
    {
        const int totalPairs = (N / 16) * (OUT_ / 32);     // 6250
        int blocks = (totalPairs + 7) / 8;
        if (blocks > 1536) blocks = 1536;
        gemm_wmma_f16<HC1, OUT_><<<blocks, 256, 0, stream>>>(agg1, W2h, h2, totalPairs);
    }
    attn_scores<<<N, OUT_, 0, stream>>>(h2, a_src2, a_dst2, als2, ald2, 1, OUT_);

    fill_kernel<<<2048, 256, 0, stream>>>(m2, -INFINITY, N);
    fill_kernel<<<2048, 256, 0, stream>>>(dn2, 0.0f, N);
    fill_kernel<<<4096, 256, 0, stream>>>(out, 0.0f, N * OUT_);

    edge_logits_max<<<(EA + 255) / 256, 256, 0, stream>>>(ei, E, EA, als2, ald2, eb2, m2, 1);
    edge_expsum    <<<(EA + 255) / 256, 256, 0, stream>>>(ei, E, EA, eb2, m2, dn2, 1);
    edge_aggregate <<<EA, OUT_, 0, stream>>>(ei, E, h2, eb2, dn2, out, 1, OUT_);

    add_bias<<<2048, 256, 0, stream>>>(out, b2, OUT_, N * OUT_);
}
